// SimpleSelfAttention_16071767622058
// MI455X (gfx1250) — matrix operand order repaired and checked by reading, not compiled, not run on hardware
//
#include <hip/hip_runtime.h>
#include <hip/hip_bf16.h>

// ---------------------------------------------------------------------------
// SimpleSelfAttention on MI455X (gfx1250, wave32).
//
// out[b] = gamma * (X X^T) (W/sigma) X + X        X: [64, 16384] per batch
// Rewritten: out[b] = M_b X + X with M_b = gamma * (G_b Wsn),  G_b = X X^T.
//
// Precision: split-bf16 (hi+lo residual) with f32 accumulate;
// 3x v_wmma_f32_16x16x32_bf16 per product tile (hi*hi + hi*lo + lo*hi).
// LDS holds pair-packed bf16 fragments (two adjacent-K values per u32) in
// separate hi/lo arrays, so fragment assembly is pure LDS loads (no VALU); b128 except out_kernel's X (b32).
// Double-buffered LDS staging: one barrier per K-step, loads overlap compute.
// ---------------------------------------------------------------------------

#define CDIM   64
#define NDIM   16384           // 128*128
#define BATCH  32
#define NCHUNK 8               // gram reduction split over N
#define CHUNK  (NDIM / NCHUNK) // 2048
#define KSTEP  64              // gram K tile per stage
#define NSTEP  (CHUNK / KSTEP) // 32

typedef __attribute__((ext_vector_type(16))) __bf16 v16bf;
typedef __attribute__((ext_vector_type(8)))  float  v8f;
typedef __attribute__((ext_vector_type(2)))  float  v2f;
typedef __attribute__((ext_vector_type(2)))  __bf16 v2bf;

struct __align__(16) Quad { unsigned int w[4]; };

union Frag {
    v16bf v;
    unsigned int u[8];
    Quad q[2];
};

__device__ __forceinline__ v8f wmma_bf16(v16bf a, v16bf b, v8f c) {
    // (neg_a, A, neg_b, B, c_mod, C, reuse_a, reuse_b)
    return __builtin_amdgcn_wmma_f32_16x16x32_bf16(false, a, false, b,
                                                   (short)0, c, false, false);
}

// Two adjacent-K floats -> pair-packed bf16 hi word + bf16 residual word.
// Expressed as vector converts so the backend can select v_cvt_pk_bf16_f32.
__device__ __forceinline__ void pack2(float f0, float f1,
                                      unsigned int& h, unsigned int& l) {
    v2f f = {f0, f1};
    v2bf hv = __builtin_convertvector(f, v2bf);          // RNE narrow
    v2f hf = __builtin_convertvector(hv, v2f);           // exact widen
    v2f r = f - hf;
    v2bf lv = __builtin_convertvector(r, v2bf);
    h = __builtin_bit_cast(unsigned int, hv);
    l = __builtin_bit_cast(unsigned int, lv);
}

// ---------------------------------------------------------------------------
// Kernel 1: spectral norm (one power-iteration step) -> Wsn = W / sigma
// ---------------------------------------------------------------------------
__global__ __launch_bounds__(64) void sn_kernel(const float* __restrict__ W,
                                                const float* __restrict__ u,
                                                float* __restrict__ wsn) {
    __shared__ float sh[64];
    int t = threadIdx.x;

    // v = l2norm(W^T u)
    float vj = 0.f;
    for (int i = 0; i < 64; ++i) vj += W[i * 64 + t] * u[i];
    sh[t] = vj * vj;
    __syncthreads();
    for (int off = 32; off > 0; off >>= 1) {
        if (t < off) sh[t] += sh[t + off];
        __syncthreads();
    }
    float nv = sh[0];
    __syncthreads();
    vj /= fmaxf(sqrtf(nv), 1e-12f);
    sh[t] = vj;
    __syncthreads();

    // w = W v ; sigma = u2 . (W v) = ||Wv||^2 / max(||Wv||, eps)
    float wt = 0.f;
    for (int j = 0; j < 64; ++j) wt += W[t * 64 + j] * sh[j];
    __syncthreads();
    sh[t] = wt * wt;
    __syncthreads();
    for (int off = 32; off > 0; off >>= 1) {
        if (t < off) sh[t] += sh[t + off];
        __syncthreads();
    }
    float nw2   = sh[0];
    float sigma = nw2 / fmaxf(sqrtf(nw2), 1e-12f);
    float inv   = 1.0f / sigma;
    for (int j = 0; j < 64; ++j) wsn[t * 64 + j] = W[t * 64 + j] * inv;
}

// ---------------------------------------------------------------------------
// Kernel 2: Gram partials. Grid = BATCH * NCHUNK, 256 threads (8 waves).
// LDS: pair-packed hi/lo tiles [64 c][32 n-pairs], double buffered.
// Wave w owns tiles (rtile = w&3, coltiles = (w>>2)*2 + {0,1}).
// ---------------------------------------------------------------------------
__global__ __launch_bounds__(256) void gram_kernel(const float* __restrict__ x,
                                                   float* __restrict__ gpart) {
    __shared__ __align__(16) unsigned int th[2][64][36]; // hi pairs (+pad)
    __shared__ __align__(16) unsigned int tl[2][64][36]; // lo pairs

    int b     = blockIdx.x >> 3;
    int chunk = blockIdx.x & 7;
    int tid   = threadIdx.x;
    int lane  = tid & 31;
    int w     = tid >> 5;
    int rtile = w & 3;
    int ct0   = (w >> 2) * 2;
    int ct1   = ct0 + 1;
    int lm    = lane & 15;
    int half  = lane >> 4;
    int arow  = rtile * 16 + lm;
    int brow0 = ct0 * 16 + lm;
    int brow1 = ct1 * 16 + lm;

    const float* xb = x + (size_t)b * CDIM * NDIM + chunk * CHUNK;
    int lrow = tid >> 2;              // staging row   0..63
    int lcg  = (tid & 3) * 16;        // staging n col 0,16,32,48
    int pbase = (tid & 3) * 8;        // pair-word base in LDS row
    const float* srcbase = xb + (size_t)lrow * NDIM + lcg;

    v8f acc0 = {};
    v8f acc1 = {};
    float4 r[4];

    // ---- prologue: stage 0
    {
        const float* p = srcbase;
        __builtin_prefetch(p + KSTEP, 0, 1);
        r[0] = *(const float4*)(p + 0);
        r[1] = *(const float4*)(p + 4);
        r[2] = *(const float4*)(p + 8);
        r[3] = *(const float4*)(p + 12);
#pragma unroll
        for (int q = 0; q < 4; ++q) {
            unsigned int h0, l0, h1, l1;
            pack2(r[q].x, r[q].y, h0, l0);
            pack2(r[q].z, r[q].w, h1, l1);
            th[0][lrow][pbase + 2 * q]     = h0;
            tl[0][lrow][pbase + 2 * q]     = l0;
            th[0][lrow][pbase + 2 * q + 1] = h1;
            tl[0][lrow][pbase + 2 * q + 1] = l1;
        }
    }
    __syncthreads();

    for (int s = 0; s < NSTEP; ++s) {
        int buf = s & 1;
        if (s + 1 < NSTEP) {
            const float* p = srcbase + (s + 1) * KSTEP;
            __builtin_prefetch(p + KSTEP, 0, 1);
            r[0] = *(const float4*)(p + 0);
            r[1] = *(const float4*)(p + 4);
            r[2] = *(const float4*)(p + 8);
            r[3] = *(const float4*)(p + 12);
        }

#pragma unroll
        for (int ksu = 0; ksu < 2; ++ksu) {
            Frag Ah, Al, Bh0, Bl0, Bh1, Bl1;
            {
                const Quad* rh = (const Quad*)&th[buf][arow][0];
                const Quad* rl = (const Quad*)&tl[buf][arow][0];
                int pa = ksu * 4 + half * 2;       // quad index
                Ah.q[0] = rh[pa];  Ah.q[1] = rh[pa + 1];
                Al.q[0] = rl[pa];  Al.q[1] = rl[pa + 1];
            }
            {
                const Quad* rh = (const Quad*)&th[buf][brow0][0];
                const Quad* rl = (const Quad*)&tl[buf][brow0][0];
                int pb = ksu * 4 + half * 2;       // quad index
                Bh0.q[0] = rh[pb]; Bh0.q[1] = rh[pb + 1];
                Bl0.q[0] = rl[pb]; Bl0.q[1] = rl[pb + 1];
            }
            {
                const Quad* rh = (const Quad*)&th[buf][brow1][0];
                const Quad* rl = (const Quad*)&tl[buf][brow1][0];
                int pb = ksu * 4 + half * 2;
                Bh1.q[0] = rh[pb]; Bh1.q[1] = rh[pb + 1];
                Bl1.q[0] = rl[pb]; Bl1.q[1] = rl[pb + 1];
            }
            acc0 = wmma_bf16(Ah.v, Bh0.v, acc0);
            acc1 = wmma_bf16(Ah.v, Bh1.v, acc1);
            acc0 = wmma_bf16(Ah.v, Bl0.v, acc0);
            acc1 = wmma_bf16(Ah.v, Bl1.v, acc1);
            acc0 = wmma_bf16(Al.v, Bh0.v, acc0);
            acc1 = wmma_bf16(Al.v, Bh1.v, acc1);
        }

        if (s + 1 < NSTEP) {
            int nbuf = buf ^ 1;
#pragma unroll
            for (int q = 0; q < 4; ++q) {
                unsigned int h0, l0, h1, l1;
                pack2(r[q].x, r[q].y, h0, l0);
                pack2(r[q].z, r[q].w, h1, l1);
                th[nbuf][lrow][pbase + 2 * q]     = h0;
                tl[nbuf][lrow][pbase + 2 * q]     = l0;
                th[nbuf][lrow][pbase + 2 * q + 1] = h1;
                tl[nbuf][lrow][pbase + 2 * q + 1] = l1;
            }
        }
        __syncthreads();
    }

    // D layout: VGPR v -> row m = v + 8*half; col = lm.
    float* gp = gpart + ((size_t)b * NCHUNK + chunk) * (CDIM * CDIM);
#pragma unroll
    for (int v = 0; v < 8; ++v) {
        int row = rtile * 16 + v + 8 * half;
        gp[row * CDIM + ct0 * 16 + lm] = acc0[v];
        gp[row * CDIM + ct1 * 16 + lm] = acc1[v];
    }
}

// ---------------------------------------------------------------------------
// Kernel 3: M_b = gamma * (G_b @ Wsn). Grid = BATCH, 64 threads.
// ---------------------------------------------------------------------------
__global__ __launch_bounds__(64) void mk_kernel(const float* __restrict__ gpart,
                                                const float* __restrict__ wsn,
                                                const float* __restrict__ gamma,
                                                float* __restrict__ Mmat) {
    __shared__ float Gsm[64][65];
    __shared__ float Wsm[64][64];
    int b = blockIdx.x;
    int t = threadIdx.x;

    for (int e = t; e < 4096; e += 64) {
        float s = 0.f;
        for (int ch = 0; ch < NCHUNK; ++ch)
            s += gpart[((size_t)b * NCHUNK + ch) * 4096 + e];
        Gsm[e >> 6][e & 63] = s;
    }
    for (int e = t; e < 4096; e += 64) Wsm[e >> 6][e & 63] = wsn[e];
    __syncthreads();

    float g = gamma[0];
    float* mb = Mmat + (size_t)b * 4096 + t * 64;
    for (int j = 0; j < 64; ++j) {
        float s = 0.f;
        for (int k = 0; k < 64; ++k) s += Gsm[t][k] * Wsm[k][j];
        mb[j] = g * s;
    }
}

// ---------------------------------------------------------------------------
// Kernel 4: out = M_b @ X + X. Grid = BATCH * 8 slabs, 256 threads (8 waves).
// Stage = 64(c) x 128(n) tile of X, pair-packed along C, double buffered.
// Wave w computes column group g = w (16 n) for all four 16-row tiles.
// ---------------------------------------------------------------------------
__global__ __launch_bounds__(256) void out_kernel(const float* __restrict__ x,
                                                  const float* __restrict__ Mmat,
                                                  float* __restrict__ out) {
    __shared__ __align__(16) unsigned int xh[2][32][132]; // c-pair x n, hi
    __shared__ __align__(16) unsigned int xl[2][32][132]; // c-pair x n, lo
    __shared__ __align__(16) unsigned int mh[64][36];     // M row x k-pair, hi
    __shared__ __align__(16) unsigned int ml[64][36];     // lo

    int b    = blockIdx.x >> 3;
    int slab = blockIdx.x & 7;
    int tid  = threadIdx.x;
    int lane = tid & 31;
    int w    = tid >> 5;
    int lm   = lane & 15;
    int half = lane >> 4;
    int ncol = w * 16 + lm;

    const float* xb = x + (size_t)b * CDIM * NDIM;
    float* ob       = out + (size_t)b * CDIM * NDIM;
    int nslab       = slab * 2048;

    // ---- stage M (pair-packed along k = c_in)
    {
        const float* Mb = Mmat + (size_t)b * 4096;
        int mrow = tid >> 2;
        int mcg  = (tid & 3) * 16;
        int mpb  = (tid & 3) * 8;
        const float* p = Mb + mrow * 64 + mcg;
#pragma unroll
        for (int q = 0; q < 4; ++q) {
            float4 a = *(const float4*)(p + 4 * q);
            unsigned int h0, l0, h1, l1;
            pack2(a.x, a.y, h0, l0);
            pack2(a.z, a.w, h1, l1);
            mh[mrow][mpb + 2 * q]     = h0;
            ml[mrow][mpb + 2 * q]     = l0;
            mh[mrow][mpb + 2 * q + 1] = h1;
            ml[mrow][mpb + 2 * q + 1] = l1;
        }
    }

    // ---- x staging geometry: thread owns c-pair p = tid>>3, 16 n columns.
    int sp  = tid >> 3;               // 0..31  (c rows 2sp, 2sp+1)
    int ncb = (tid & 7) * 16;         // n col base
    const float* rowA = xb + (size_t)(2 * sp) * NDIM + nslab + ncb;
    const float* rowB = rowA + NDIM;

    float4 ra[4], rb[4];
    {
        __builtin_prefetch(rowA + 128, 0, 1);
        __builtin_prefetch(rowB + 128, 0, 1);
#pragma unroll
        for (int q = 0; q < 4; ++q) {
            ra[q] = *(const float4*)(rowA + 4 * q);
            rb[q] = *(const float4*)(rowB + 4 * q);
        }
#pragma unroll
        for (int q = 0; q < 4; ++q) {
            Quad qh, ql;
            pack2(ra[q].x, rb[q].x, qh.w[0], ql.w[0]);
            pack2(ra[q].y, rb[q].y, qh.w[1], ql.w[1]);
            pack2(ra[q].z, rb[q].z, qh.w[2], ql.w[2]);
            pack2(ra[q].w, rb[q].w, qh.w[3], ql.w[3]);
            *(Quad*)&xh[0][sp][ncb + 4 * q] = qh;
            *(Quad*)&xl[0][sp][ncb + 4 * q] = ql;
        }
    }
    __syncthreads();

    for (int st = 0; st < 16; ++st) {
        int buf = st & 1;
        int nb  = nslab + st * 128;

        if (st + 1 < 16) {
            const float* pA = rowA + (st + 1) * 128;
            const float* pB = rowB + (st + 1) * 128;
            __builtin_prefetch(pA + 128, 0, 1);
            __builtin_prefetch(pB + 128, 0, 1);
#pragma unroll
            for (int q = 0; q < 4; ++q) {
                ra[q] = *(const float4*)(pA + 4 * q);
                rb[q] = *(const float4*)(pB + 4 * q);
            }
        }

        // B fragments for this wave's column group.
        Frag Bh[2], Bl[2];
#pragma unroll
        for (int ksu = 0; ksu < 2; ++ksu) {
#pragma unroll
            for (int v = 0; v < 8; ++v) {
                int pk = ksu * 16 + half * 8 + v;   // c-pair row
                Bh[ksu].u[v] = xh[buf][pk][ncol];
                Bl[ksu].u[v] = xl[buf][pk][ncol];
            }
        }

#pragma unroll
        for (int rt = 0; rt < 4; ++rt) {
            v8f acc = {};
            int arow = rt * 16 + lm;
#pragma unroll
            for (int ksu = 0; ksu < 2; ++ksu) {
                Frag Ah, Al;
                const Quad* rh = (const Quad*)&mh[arow][0];
                const Quad* rl = (const Quad*)&ml[arow][0];
                int pa = ksu * 4 + half * 2;
                Ah.q[0] = rh[pa];  Ah.q[1] = rh[pa + 1];
                Al.q[0] = rl[pa];  Al.q[1] = rl[pa + 1];
                acc = wmma_bf16(Ah.v, Bh[ksu].v, acc);
                acc = wmma_bf16(Ah.v, Bl[ksu].v, acc);
                acc = wmma_bf16(Al.v, Bh[ksu].v, acc);
            }
            // writeback: D VGPR v -> row m = v + 8*half, col = lm.
#pragma unroll
            for (int v = 0; v < 8; ++v) {
                int crow   = rt * 16 + v + 8 * half;
                size_t idx = (size_t)crow * NDIM + nb + w * 16 + lm;
                ob[idx] = acc[v] + xb[idx];
            }
        }

        if (st + 1 < 16) {
            int nbuf = buf ^ 1;
#pragma unroll
            for (int q = 0; q < 4; ++q) {
                Quad qh, ql;
                pack2(ra[q].x, rb[q].x, qh.w[0], ql.w[0]);
                pack2(ra[q].y, rb[q].y, qh.w[1], ql.w[1]);
                pack2(ra[q].z, rb[q].z, qh.w[2], ql.w[2]);
                pack2(ra[q].w, rb[q].w, qh.w[3], ql.w[3]);
                *(Quad*)&xh[nbuf][sp][ncb + 4 * q] = qh;
                *(Quad*)&xl[nbuf][sp][ncb + 4 * q] = ql;
            }
        }
        __syncthreads();
    }
}

// ---------------------------------------------------------------------------
extern "C" void kernel_launch(void* const* d_in, const int* in_sizes, int n_in,
                              void* d_out, int out_size, void* d_ws, size_t ws_size,
                              hipStream_t stream) {
    (void)in_sizes; (void)n_in; (void)out_size; (void)ws_size;

    const float* x     = (const float*)d_in[0];   // [32,64,128,128]
    const float* W     = (const float*)d_in[1];   // [64,64]
    const float* gamma = (const float*)d_in[2];   // [1]
    const float* u     = (const float*)d_in[3];   // [64]
    float* out         = (float*)d_out;

    // Workspace layout (floats):
    float* wsn   = (float*)d_ws;          // 4096
    float* Mmat  = wsn + 4096;            // 32 * 4096
    float* gpart = Mmat + 32 * 4096;      // 32 * 8 * 4096  (~4 MiB)

    sn_kernel<<<1, 64, 0, stream>>>(W, u, wsn);
    gram_kernel<<<BATCH * NCHUNK, 256, 0, stream>>>(x, gpart);
    mk_kernel<<<BATCH, 64, 0, stream>>>(gpart, wsn, gamma, Mmat);
    out_kernel<<<BATCH * 8, 256, 0, stream>>>(x, Mmat, out);
}
